// NestedEventMamba_57415122813611
// MI455X (gfx1250) — matrix operand order repaired and checked by reading, not compiled, not run on hardware
//
#include <hip/hip_runtime.h>
#include <hip/hip_bf16.h>

// ---------------------------------------------------------------------------
// MI455X (gfx1250) nested-Mamba forward.
// wave32; every GEMM on v_wmma_f32_16x16x32_f16 with vectorized b128 fragment
// loads (weights pre-transposed so B is contiguous per lane), 4 N-tiles per
// wave for A-fragment reuse, scalar (readfirstlane) tile control flow so EXEC
// stays all-ones via plain s_cbranch; LDS-staged SSM scan.
// ---------------------------------------------------------------------------

typedef __attribute__((ext_vector_type(16))) _Float16 v16h;
typedef __attribute__((ext_vector_type(8)))  _Float16 v8h;
typedef __attribute__((ext_vector_type(8)))  float    v8f;

__device__ __forceinline__ float geluf(float x) {
    return 0.5f * x * (1.0f + erff(x * 0.70710678118654752440f));
}
__device__ __forceinline__ float siluf(float x) {
    return x / (1.0f + expf(-x));
}
__device__ __forceinline__ float softplusf(float x) {
    return (x > 20.0f) ? x : log1pf(expf(x));
}

// ---------------------------------------------------------------------------
// WMMA GEMM: out = act(A[M,K]f16 @ W[K,N] + bias) (+ resid)
// W supplied TRANSPOSED as Wt[N, Kp]: each lane's B fragment is two runs of 8 contiguous
// halves (two b128 loads), mirroring the A side.
// Each wave computes a 16-row x 64-col strip (4 accumulators); tile indices
// are forced scalar via readfirstlane so per-tile guards are s_cbranch (no
// EXEC churn around the WMMAs).
//
// Fragment layouts (16-bit WMMA, wave32):
//   A: lane&15 = M row, lane>>4 = K-half; halves j=0..7 -> K=k0+lh*8+j,
//      j=8..15 -> K=k0+16+lh*8+(j-8)  => two contiguous 8-half loads.
//   B: lane&15 = N col, lane>>4 = K-half; same K mapping as A
//      => two contiguous 8-half loads of Wt row `col`.
//   C: VGPR r -> M = r + 8*(lane>>4), N col = lane&15.
// ---------------------------------------------------------------------------
__global__ __launch_bounds__(256)
void gemm_wmma(const _Float16* __restrict__ A, const _Float16* __restrict__ Wt,
               const float* __restrict__ bias, const float* __restrict__ resid,
               float* __restrict__ outF, _Float16* __restrict__ outH,
               int M, int N, int K, int lda, int ldwt, int ldf, int ldh, int act)
{
    const int lane = threadIdx.x & 31;
    // wave index is uniform within the wave: pin it to an SGPR so all tile
    // arithmetic and guards below are provably scalar.
    const int wave = __builtin_amdgcn_readfirstlane(threadIdx.x >> 5);
    const int ntiles  = (N + 15) >> 4;
    const int ngroups = (ntiles + 3) >> 2;       // 4 N-tiles per wave
    const int mtiles  = M >> 4;
    const int grp = blockIdx.x * 8 + wave;
    if (grp >= ngroups * mtiles) return;          // scalar-uniform exit
    const int tm  = grp / ngroups;
    const int tg  = grp - tm * ngroups;
    const int tn0 = tg * 4;
    const bool v1 = (tn0 + 1) < ntiles;           // scalar predicates
    const bool v2 = (tn0 + 2) < ntiles;
    const bool v3 = (tn0 + 3) < ntiles;

    const int lo16 = lane & 15;
    const int lh   = lane >> 4;
    const int row  = (tm << 4) + lo16;

    union AV { v16h v; v8h h8[2]; _Float16 e[16]; };
    AV a, b;
    v8f c0 = {}, c1 = {}, c2 = {}, c3 = {};

    const _Float16* ap  = A + (size_t)row * lda + lh * 8;
    const _Float16* bp0 = Wt + (size_t)(tn0 * 16 + lo16) * ldwt + lh * 8;
    const size_t bstride = (size_t)16 * ldwt;     // one N-tile of Wt rows

    for (int k0 = 0; k0 < K; k0 += 32) {
        a.h8[0] = *(const v8h*)(ap);
        a.h8[1] = *(const v8h*)(ap + 16);
        __builtin_prefetch((const void*)(ap + 32), 0, 3);   // near prefetch
        ap += 32;

        b.h8[0] = *(const v8h*)(bp0);
        b.h8[1] = *(const v8h*)(bp0 + 16);
        c0 = __builtin_amdgcn_wmma_f32_16x16x32_f16(false, a.v, false, b.v,
                                                    (short)0, c0, false, false);
        if (v1) {
            b.h8[0] = *(const v8h*)(bp0 + bstride);
            b.h8[1] = *(const v8h*)(bp0 + bstride + 16);
            c1 = __builtin_amdgcn_wmma_f32_16x16x32_f16(false, a.v, false, b.v,
                                                        (short)0, c1, false, false);
        }
        if (v2) {
            b.h8[0] = *(const v8h*)(bp0 + 2 * bstride);
            b.h8[1] = *(const v8h*)(bp0 + 2 * bstride + 16);
            c2 = __builtin_amdgcn_wmma_f32_16x16x32_f16(false, a.v, false, b.v,
                                                        (short)0, c2, false, false);
        }
        if (v3) {
            b.h8[0] = *(const v8h*)(bp0 + 3 * bstride);
            b.h8[1] = *(const v8h*)(bp0 + 3 * bstride + 16);
            c3 = __builtin_amdgcn_wmma_f32_16x16x32_f16(false, a.v, false, b.v,
                                                        (short)0, c3, false, false);
        }
        bp0 += 32;
    }

    const int mbase = (tm << 4) + 8 * lh;
    v8f acc[4] = { c0, c1, c2, c3 };
    #pragma unroll
    for (int nt = 0; nt < 4; ++nt) {
        if (nt == 1 && !v1) break;                // scalar guards
        if (nt == 2 && !v2) break;
        if (nt == 3 && !v3) break;
        int col = (tn0 + nt) * 16 + lo16;
        if (col < N) {
            #pragma unroll
            for (int r = 0; r < 8; ++r) {
                int m = mbase + r;
                float v = acc[nt][r];
                if (bias) v += bias[col];
                if (act == 1) v = geluf(v);
                if (resid) v += resid[(size_t)m * ldf + col];
                if (outF) outF[(size_t)m * ldf + col] = v;
                if (outH) outH[(size_t)m * ldh + col] = (_Float16)v;
            }
        }
    }
}

// ---------------------------------------------------------------------------
// LayerNorm over D=128 (one token per 128-thread block)
// ---------------------------------------------------------------------------
__global__ __launch_bounds__(128)
void ln_kernel(const float* __restrict__ x, const float* __restrict__ g,
               const float* __restrict__ b, float* __restrict__ outF,
               _Float16* __restrict__ outH)
{
    __shared__ float red[128];
    const int t = blockIdx.x, d = threadIdx.x;
    float v = x[(size_t)t * 128 + d];
    red[d] = v; __syncthreads();
    for (int s = 64; s > 0; s >>= 1) { if (d < s) red[d] += red[d + s]; __syncthreads(); }
    float mean = red[0] * (1.0f / 128.0f); __syncthreads();
    float dv = v - mean;
    red[d] = dv * dv; __syncthreads();
    for (int s = 64; s > 0; s >>= 1) { if (d < s) red[d] += red[d + s]; __syncthreads(); }
    float var = red[0] * (1.0f / 128.0f);
    float y = dv * rsqrtf(var + 1e-5f) * g[d] + b[d];
    if (outF) outF[(size_t)t * 128 + d] = y;
    if (outH) outH[(size_t)t * 128 + d] = (_Float16)y;
}

// ---------------------------------------------------------------------------
// block-level depthwise conv (same padding) + lc_b + residual xn -> u (f16)
// ---------------------------------------------------------------------------
__global__ __launch_bounds__(128)
void lcconv_kernel(const float* __restrict__ xn, const float* __restrict__ w,
                   const float* __restrict__ b, _Float16* __restrict__ u16,
                   int Lseq, int ksz, int pad)
{
    const int t = blockIdx.x, d = threadIdx.x;
    const int s = t / Lseq, l = t - s * Lseq;
    float acc = b[d];
    for (int j = 0; j < ksz; ++j) {
        int ll = l - pad + j;
        if (ll >= 0 && ll < Lseq)
            acc += xn[(size_t)(s * Lseq + ll) * 128 + d] * w[d * ksz + j];
    }
    acc += xn[(size_t)t * 128 + d];
    u16[(size_t)t * 128 + d] = (_Float16)acc;
}

// ---------------------------------------------------------------------------
// mamba causal depthwise conv (K=4, pad (3,0)) + silu ; xi = xz[:, :256]
// ---------------------------------------------------------------------------
__global__ __launch_bounds__(256)
void mconv_kernel(const float* __restrict__ xz, const float* __restrict__ cw,
                  const float* __restrict__ cb, float* __restrict__ xc,
                  _Float16* __restrict__ xc16, int Lseq)
{
    const int t = blockIdx.x, d = threadIdx.x;
    const int s = t / Lseq, l = t - s * Lseq;
    float acc = cb[d];
    #pragma unroll
    for (int j = 0; j < 4; ++j) {
        int ll = l - 3 + j;
        if (ll >= 0)
            acc += xz[(size_t)(s * Lseq + ll) * 512 + d] * cw[d * 4 + j];
    }
    acc = siluf(acc);
    xc[(size_t)t * 256 + d]   = acc;
    xc16[(size_t)t * 256 + d] = (_Float16)acc;
}

// ---------------------------------------------------------------------------
// Selective-scan: one workgroup per sequence, 256 threads = Di channels.
// dbc row (72 f32) staged in LDS each timestep; h[32], exp(A_log)[32] in regs.
// ---------------------------------------------------------------------------
__global__ __launch_bounds__(256)
void scan_kernel(const float* __restrict__ dbc, const float* __restrict__ xc,
                 const float* __restrict__ xz, const float* __restrict__ dt_w,
                 const float* __restrict__ dt_b, const float* __restrict__ A_log,
                 const float* __restrict__ Dp, _Float16* __restrict__ y16, int Lseq)
{
    __shared__ float sd[72];
    const int s = blockIdx.x, d = threadIdx.x;
    float dtw[8];
    #pragma unroll
    for (int q = 0; q < 8; ++q) dtw[q] = dt_w[q * 256 + d];
    float eA[32], h[32];
    #pragma unroll
    for (int n = 0; n < 32; ++n) { eA[n] = expf(A_log[d * 32 + n]); h[n] = 0.0f; }
    const float dtb = dt_b[d], Dv = Dp[d];

    for (int l = 0; l < Lseq; ++l) {
        const int t = s * Lseq + l;
        __syncthreads();
        if (d < 72) sd[d] = dbc[(size_t)t * 72 + d];
        __syncthreads();
        float dtr = dtb;
        #pragma unroll
        for (int q = 0; q < 8; ++q) dtr += sd[q] * dtw[q];
        const float dt  = softplusf(dtr);
        const float xcv = xc[(size_t)t * 256 + d];
        const float zv  = xz[(size_t)t * 512 + 256 + d];
        float y = 0.0f;
        #pragma unroll
        for (int n = 0; n < 32; ++n) {
            float dA = expf(-dt * eA[n]);
            h[n] = dA * h[n] + (dt * sd[8 + n]) * xcv;
            y += h[n] * sd[40 + n];
        }
        float yv = (y + xcv * Dv) * siluf(zv);
        y16[(size_t)t * 256 + d] = (_Float16)yv;
    }
}

// ---------------------------------------------------------------------------
// (mean + max)/2 over L per (sequence, feature)
// ---------------------------------------------------------------------------
__global__ __launch_bounds__(128)
void pool_kernel(const float* __restrict__ x, float* __restrict__ seq, int Lseq)
{
    const int s = blockIdx.x, d = threadIdx.x;
    float sum = 0.0f, mx = -3.4e38f;
    for (int l = 0; l < Lseq; ++l) {
        float v = x[(size_t)(s * Lseq + l) * 128 + d];
        sum += v; mx = fmaxf(mx, v);
    }
    seq[(size_t)s * 128 + d] = 0.5f * (sum / (float)Lseq + mx);
}

// ---------------------------------------------------------------------------
// Final head: mean over W, LN, 128->64 gelu, 64->11. M=2 -> plain VALU.
// ---------------------------------------------------------------------------
__global__ __launch_bounds__(128)
void head_kernel(const float* __restrict__ seq, const float* __restrict__ hg,
                 const float* __restrict__ hb, const float* __restrict__ w1,
                 const float* __restrict__ b1, const float* __restrict__ w2,
                 const float* __restrict__ b2, float* __restrict__ out)
{
    __shared__ float red[128];
    __shared__ float gn[128];
    __shared__ float hh[64];
    const int d = threadIdx.x;
    for (int bq = 0; bq < 2; ++bq) {
        float g = 0.0f;
        for (int w = 0; w < 16; ++w) g += seq[(size_t)(bq * 16 + w) * 128 + d];
        g *= (1.0f / 16.0f);
        red[d] = g; __syncthreads();
        for (int s = 64; s > 0; s >>= 1) { if (d < s) red[d] += red[d + s]; __syncthreads(); }
        float mean = red[0] * (1.0f / 128.0f); __syncthreads();
        float dv = g - mean;
        red[d] = dv * dv; __syncthreads();
        for (int s = 64; s > 0; s >>= 1) { if (d < s) red[d] += red[d + s]; __syncthreads(); }
        float var = red[0] * (1.0f / 128.0f); __syncthreads();
        gn[d] = dv * rsqrtf(var + 1e-5f) * hg[d] + hb[d];
        __syncthreads();
        if (d < 64) {
            float acc = b1[d];
            for (int j = 0; j < 128; ++j) acc += gn[j] * w1[j * 64 + d];
            hh[d] = geluf(acc);
        }
        __syncthreads();
        if (d < 11) {
            float acc = b2[d];
            for (int j = 0; j < 64; ++j) acc += hh[j] * w2[j * 11 + d];
            out[bq * 11 + d] = acc;
        }
        __syncthreads();
    }
}

// ---------------------------------------------------------------------------
// f32 [K,N] -> zero-padded TRANSPOSED f16 [Np, Kp]  (Wt[n][k] = W[k][n])
// ---------------------------------------------------------------------------
__global__ void pack_w16t(const float* __restrict__ src, _Float16* __restrict__ dst,
                          int K, int N, int Kp, int Np)
{
    int idx = blockIdx.x * 256 + threadIdx.x;
    if (idx >= Kp * Np) return;
    int n = idx / Kp, k = idx - n * Kp;
    dst[idx] = (k < K && n < N) ? (_Float16)src[k * N + n] : (_Float16)0.0f;
}

// concat [z_real|z_imag|coords] -> f16 [4096, 160] (K padded 130 -> 160)
__global__ void pack_feat(const float* __restrict__ zr, const float* __restrict__ zi,
                          const float* __restrict__ co, _Float16* __restrict__ f16)
{
    int idx = blockIdx.x * 256 + threadIdx.x;
    if (idx >= 4096 * 160) return;
    int t = idx / 160, i = idx - t * 160;
    float v = 0.0f;
    if (i < 64)       v = zr[t * 64 + i];
    else if (i < 128) v = zi[t * 64 + (i - 64)];
    else if (i < 130) v = co[t * 2 + (i - 128)];
    f16[idx] = (_Float16)v;
}

// ---------------------------------------------------------------------------
extern "C" void kernel_launch(void* const* d_in, const int* in_sizes, int n_in,
                              void* d_out, int out_size, void* d_ws, size_t ws_size,
                              hipStream_t stream)
{
    (void)in_sizes; (void)n_in; (void)out_size; (void)ws_size;

    // ---- input walk (setup_inputs dict order, recursively flattened) ----
    int p = 0;
    auto F = [&]() -> const float* { return (const float*)d_in[p++]; };
    const float* z_real = F(); const float* z_imag = F(); const float* coords = F();
    const float* ad_w1 = F(); const float* ad_b1 = F();
    const float* ad_w2 = F(); const float* ad_b2 = F();
    struct Blk {
        const float *n1g,*n1b,*lcw,*lcb;
        const float *inw,*cw,*cb,*xw,*dtw,*dtb,*alog,*Dp,*outw;
        const float *n2g,*n2b,*m1w,*m1b,*m2w,*m2b;
    };
    Blk bk[4];
    for (int b = 0; b < 4; ++b) {
        bk[b].n1g = F(); bk[b].n1b = F(); bk[b].lcw = F(); bk[b].lcb = F();
        bk[b].inw = F(); bk[b].cw  = F(); bk[b].cb  = F(); bk[b].xw  = F();
        bk[b].dtw = F(); bk[b].dtb = F(); bk[b].alog= F(); bk[b].Dp  = F();
        bk[b].outw= F();
        bk[b].n2g = F(); bk[b].n2b = F(); bk[b].m1w = F(); bk[b].m1b = F();
        bk[b].m2w = F(); bk[b].m2b = F();
    }
    const float* h_g  = F(); const float* h_b  = F();
    const float* h_w1 = F(); const float* h_b1 = F();
    const float* h_w2 = F(); const float* h_b2 = F();

    // ---- workspace arena ----
    char* wsb = (char*)d_ws; size_t off = 0;
    auto take = [&](size_t bytes) -> char* {
        char* r = wsb + off;
        off += (bytes + 255) & ~(size_t)255;
        return r;
    };
    // transposed f16 weights: Wt[Np][Kp]
    _Float16* w_ad1 = (_Float16*)take((size_t)128 * 160 * 2);
    _Float16* w_ad2 = (_Float16*)take((size_t)128 * 128 * 2);
    _Float16 *w_in[4], *w_xw[4], *w_ow[4], *w_m1[4], *w_m2[4];
    for (int b = 0; b < 4; ++b) {
        w_in[b] = (_Float16*)take((size_t)512 * 128 * 2);
        w_xw[b] = (_Float16*)take((size_t)80  * 256 * 2);
        w_ow[b] = (_Float16*)take((size_t)128 * 256 * 2);
        w_m1[b] = (_Float16*)take((size_t)512 * 128 * 2);
        w_m2[b] = (_Float16*)take((size_t)128 * 512 * 2);
    }
    _Float16* feat16 = (_Float16*)take((size_t)4096 * 160 * 2);
    _Float16* t116   = (_Float16*)take((size_t)4096 * 128 * 2);
    float*    xbuf   = (float*)   take((size_t)4096 * 128 * 4);
    float*    xnbuf  = (float*)   take((size_t)4096 * 128 * 4);
    _Float16* u16    = (_Float16*)take((size_t)4096 * 128 * 2);
    float*    xz     = (float*)   take((size_t)4096 * 512 * 4);
    float*    xc     = (float*)   take((size_t)4096 * 256 * 4);
    _Float16* xc16   = (_Float16*)take((size_t)4096 * 256 * 2);
    float*    dbc    = (float*)   take((size_t)4096 * 72  * 4);
    _Float16* y16    = (_Float16*)take((size_t)4096 * 256 * 2);
    _Float16* xn216  = (_Float16*)take((size_t)4096 * 128 * 2);
    _Float16* h16    = (_Float16*)take((size_t)4096 * 512 * 2);
    float*    seqb   = (float*)   take((size_t)32 * 128 * 4);

    // ---- helpers ----
    auto P = [&](const float* s, _Float16* d, int K, int N, int Kp, int Np) {
        int n = Kp * Np;
        pack_w16t<<<(n + 255) / 256, 256, 0, stream>>>(s, d, K, N, Kp, Np);
    };
    auto G = [&](const _Float16* A, const _Float16* Wt, const float* bias,
                 const float* resid, float* oF, _Float16* oH,
                 int M, int N, int Kp, int lda, int ldf, int ldh, int act) {
        int ntiles = (N + 15) / 16;
        int groups = ((ntiles + 3) / 4) * (M / 16);
        gemm_wmma<<<(groups + 7) / 8, 256, 0, stream>>>(
            A, Wt, bias, resid, oF, oH, M, N, Kp, lda, /*ldwt=*/Kp, ldf, ldh, act);
    };

    // ---- weight conversion (deterministic, every call) ----
    P(ad_w1, w_ad1, 130, 128, 160, 128);
    P(ad_w2, w_ad2, 128, 128, 128, 128);
    for (int b = 0; b < 4; ++b) {
        P(bk[b].inw,  w_in[b], 128, 512, 128, 512);
        P(bk[b].xw,   w_xw[b], 256, 72,  256, 80);
        P(bk[b].outw, w_ow[b], 256, 128, 256, 128);
        P(bk[b].m1w,  w_m1[b], 128, 512, 128, 512);
        P(bk[b].m2w,  w_m2[b], 512, 128, 512, 128);
    }

    // ---- adapter: x = gelu(feat@ad_w1+b1)@ad_w2+b2 ----
    pack_feat<<<(4096 * 160 + 255) / 256, 256, 0, stream>>>(z_real, z_imag, coords, feat16);
    G(feat16, w_ad1, ad_b1, nullptr, nullptr, t116, 4096, 128, 160, 160, 0, 128, 1);
    G(t116,   w_ad2, ad_b2, nullptr, xbuf,    nullptr, 4096, 128, 128, 128, 128, 0, 0);

    // ---- one block (intra or inter level) ----
    auto run_block = [&](float* x, int Mtok, int Ls, int pad, int ksz, const Blk& B, int bi) {
        int nseq = Mtok / Ls;
        ln_kernel<<<Mtok, 128, 0, stream>>>(x, B.n1g, B.n1b, xnbuf, nullptr);
        lcconv_kernel<<<Mtok, 128, 0, stream>>>(xnbuf, B.lcw, B.lcb, u16, Ls, ksz, pad);
        G(u16, w_in[bi], nullptr, nullptr, xz, nullptr, Mtok, 512, 128, 128, 512, 0, 0);
        mconv_kernel<<<Mtok, 256, 0, stream>>>(xz, B.cw, B.cb, xc, xc16, Ls);
        G(xc16, w_xw[bi], nullptr, nullptr, dbc, nullptr, Mtok, 72, 256, 256, 72, 0, 0);
        scan_kernel<<<nseq, 256, 0, stream>>>(dbc, xc, xz, B.dtw, B.dtb, B.alog, B.Dp, y16, Ls);
        G(y16, w_ow[bi], nullptr, x, x, nullptr, Mtok, 128, 256, 256, 128, 0, 0);
        ln_kernel<<<Mtok, 128, 0, stream>>>(x, B.n2g, B.n2b, nullptr, xn216);
        G(xn216, w_m1[bi], B.m1b, nullptr, nullptr, h16, Mtok, 512, 128, 128, 0, 512, 1);
        G(h16,   w_m2[bi], B.m2b, x, x, nullptr, Mtok, 128, 512, 512, 128, 0, 0);
    };

    // intra level: 32 sequences of L=128, conv k=11 pad 5
    run_block(xbuf, 4096, 128, 5, 11, bk[0], 0);
    run_block(xbuf, 4096, 128, 5, 11, bk[1], 1);

    // pool over L -> inter sequence (2 batches x 16 windows)
    pool_kernel<<<32, 128, 0, stream>>>(xbuf, seqb, 128);

    // inter level: 2 sequences of L=16, conv k=3 pad 1
    run_block(seqb, 32, 16, 1, 3, bk[2], 2);
    run_block(seqb, 32, 16, 1, 3, bk[3], 3);

    // head
    head_kernel<<<1, 128, 0, stream>>>(seqb, h_g, h_b, h_w1, h_b1, h_w2, h_b2,
                                       (float*)d_out);
}
